// SelectiveSSM_44409961840736
// MI455X (gfx1250) — matrix operand order repaired and checked by reading, not compiled, not run on hardware
//
#include <hip/hip_runtime.h>
#include <hip/hip_bf16.h>
#include <math.h>

// ---------------------------------------------------------------------------
// Types (CDNA5 wave32 WMMA fragments + TDM descriptor words)
// ---------------------------------------------------------------------------
typedef __bf16 v16bf __attribute__((ext_vector_type(16)));
typedef __bf16 v8bf  __attribute__((ext_vector_type(8)));
typedef __bf16 v2bf  __attribute__((ext_vector_type(2)));
typedef float  v8f   __attribute__((ext_vector_type(8)));
typedef unsigned int v4u __attribute__((ext_vector_type(4)));
typedef int v8i __attribute__((ext_vector_type(8)));
typedef int v4i __attribute__((ext_vector_type(4)));

#define D_MODEL 1024
#define D_HID   512
#define D_STATE 16
#define SEQ     4096
#define BATCH   8
#define ROWS    (BATCH * SEQ)        // 32768 flattened (b,s) rows

// ---------------------------------------------------------------------------
// W1 pack: f32 [1024][512] -> bf16 fragment-major [kstep(32)][tile(32)][lane(32)][e(16)]
// B-fragment (32x16) of v_wmma_f32_16x16x32_bf16 = one contiguous 32B run/lane:
//   lane<16 : K = kstep*32 + e,      N = tile*16 + lane
//   lane>=16: K = kstep*32 + 16 + e, N = tile*16 + (lane-16)
// ---------------------------------------------------------------------------
__global__ __launch_bounds__(256)
void pack_w1(const float* __restrict__ W, __bf16* __restrict__ Wp) {
    int idx = blockIdx.x * 256 + threadIdx.x;         // 0 .. 1024*512-1
    int e     = idx & 15;
    int lane  = (idx >> 4) & 31;
    int t     = (idx >> 9) & 31;
    int kstep = idx >> 14;
    int k = kstep * 32 + ((lane >= 16) ? 16 : 0) + e;
    int n = t * 16 + (lane & 15);
    Wp[idx] = (__bf16)W[k * D_HID + n];
}

// W2 pack: f32 [512][NOUT] -> bf16 fragment-major [kstep(16)][lane(32)][e(16)],
// zero-padded to N=16 when NOUT<16. 8192 elements = 16 KB.
__global__ __launch_bounds__(256)
void pack_w2(const float* __restrict__ W, __bf16* __restrict__ Wp, int NOUT) {
    int idx = blockIdx.x * 256 + threadIdx.x;         // 0 .. 8191
    int e     = idx & 15;
    int lane  = (idx >> 4) & 31;
    int kstep = idx >> 9;
    int k = kstep * 32 + ((lane >= 16) ? 16 : 0) + e;
    int n = lane & 15;
    Wp[idx] = (n < NOUT) ? (__bf16)W[k * NOUT + n] : (__bf16)0.0f;
}

// ---------------------------------------------------------------------------
// TDM: 1-D 16 KB transfer (4096 dwords) global -> LDS via Tensor Data Mover.
// D# group0: count=1 | lds_addr | global_addr | type=2 ("image").
// D# group1: data_size=4B, tensor_dim0=tile_dim0=4096, tile_dim1=1.
// This toolchain's builtin is the 6-arg form:
//   (uint32x4 g0, int32x8 g1, int32x4 g2, int32x4 g3, int32x8, i32 cpol)
// ---------------------------------------------------------------------------
#if defined(__has_builtin) && __has_builtin(__builtin_amdgcn_tensor_load_to_lds)
#define HAVE_TDM 1
__device__ inline void tdm_load_16kb(const void* gsrc, unsigned lds_off) {
    unsigned long long ga = (unsigned long long)gsrc;
    v4u g0 = { 1u,                                       // count=1
               lds_off,                                  // lds_addr (bytes)
               (unsigned)(ga & 0xFFFFFFFFu),             // global_addr[31:0]
               (unsigned)(((ga >> 32) & 0x1FFFFFFu) | (2u << 30)) }; // [56:32] | type=2
    v8i g1 = { (int)(2u << 16),          // workgroup_mask=0, data_size=2 (4B)
               (int)(4096u << 16),       // tensor_dim0[15:0] (bits 63:48)
               (int)(1u << 16),          // tensor_dim0[31:16]=0 | tensor_dim1[15:0]=1
               (int)(4096u << 16),       // tensor_dim1[31:16]=0 | tile_dim0=4096
               (int)(1u),                // tile_dim1=1 | tile_dim2=0
               (int)(4096u),             // tensor_dim0_stride[31:0]
               0, 0 };
    v4i z4 = { 0, 0, 0, 0 };
    v8i z8 = { 0, 0, 0, 0, 0, 0, 0, 0 };
    __builtin_amdgcn_tensor_load_to_lds(g0, g1, z4, z4, z8, 0);
}
#else
#define HAVE_TDM 0
#endif

// ---------------------------------------------------------------------------
// Generic A-fragment loader (16-bit A 16x32, k = 16*(lane/16) + e, matching packed B) from LDS row-major.
// ---------------------------------------------------------------------------
__device__ inline v16bf load_afrag(const __bf16* base, int stride, int lane,
                                   int rtile, int kb) {
    const int row = rtile * 16 + (lane & 15);
    const int klo = (lane < 16) ? 0 : 16;
    v8bf lo = *(const v8bf*)(base + row * stride + kb + klo);
    v8bf hi = *(const v8bf*)(base + row * stride + kb + klo + 8);
    v16bf a;
#pragma unroll
    for (int i = 0; i < 8; ++i) { a[i] = lo[i]; a[i + 8] = hi[i]; }
    return a;
}

__device__ inline v16bf load_bfrag(const __bf16* p) {
    v8bf lo = *(const v8bf*)p;
    v8bf hi = *(const v8bf*)(p + 8);
    v16bf b;
#pragma unroll
    for (int i = 0; i < 8; ++i) { b[i] = lo[i]; b[i + 8] = hi[i]; }
    return b;
}

// ---------------------------------------------------------------------------
// Fused triple-MLP kernel. One workgroup = 32 rows of x; x staged to LDS
// (f32->bf16) ONCE and multiplied against all three W1's (12 WMMA / K-step /
// wave). Then, per MLP j: GELU epilogue -> hidden LDS (bf16), W2 fragment
// buffer DMA'd to LDS by the TDM, GEMM2 on tensor cores, activation epilogue
// -> gate tensors decay/Bt/Ct.
//   j=0: decay[row][n] = exp(softplus(.) * A[n][n])
//   j=1: Bt = tanh(.) * Bp[n]         j=2: Ct = sigmoid(.) * Cp[n]
// 512 threads = 16 waves; wave w owns hidden cols [w*32, w*32+32).
// ---------------------------------------------------------------------------
__global__ __launch_bounds__(512)
void mlp3_fused(const float* __restrict__ x,
                const __bf16* __restrict__ W1p0, const __bf16* __restrict__ W1p1,
                const __bf16* __restrict__ W1p2,
                const float* __restrict__ b1_0, const float* __restrict__ b1_1,
                const float* __restrict__ b1_2,
                const __bf16* __restrict__ W2p0, const __bf16* __restrict__ W2p1,
                const __bf16* __restrict__ W2p2,
                const float* __restrict__ b2_0, const float* __restrict__ b2_1,
                const float* __restrict__ b2_2,
                const float* __restrict__ auxA, const float* __restrict__ auxB,
                const float* __restrict__ auxC,
                float* __restrict__ out0, float* __restrict__ out1,
                float* __restrict__ out2)
{
    // hidden tile 32 x 520 halves (A-staging 32x40 aliases its start) + W2 stage
    __shared__ __bf16 hidden[32 * 520];
    __shared__ __bf16 w2lds[16 * 32 * 16];
    __bf16* Abuf = hidden;

    const int tid  = threadIdx.x;
    const int lane = tid & 31;
    const int wave = tid >> 5;
    const int row0 = blockIdx.x * 32;

    const __bf16* W1ps[3] = { W1p0, W1p1, W1p2 };
    const __bf16* W2ps[3] = { W2p0, W2p1, W2p2 };
    const float*  b1s[3]  = { b1_0, b1_1, b1_2 };
    const float*  b2s[3]  = { b2_0, b2_1, b2_2 };
    const float*  auxs[3] = { auxA, auxB, auxC };
    float*        outs[3] = { out0, out1, out2 };

    v8f acc[3][2][2];
#pragma unroll
    for (int j = 0; j < 3; ++j)
#pragma unroll
        for (int r = 0; r < 2; ++r)
#pragma unroll
            for (int c = 0; c < 2; ++c)
#pragma unroll
                for (int i = 0; i < 8; ++i) acc[j][r][c][i] = 0.0f;

    // ============================ GEMM1 =================================
    for (int kk = 0; kk < D_MODEL; kk += 32) {
        // stage x tile (32 rows x 32 k) f32 -> bf16 LDS; 512 thr x 2 elems
        {
            const int r  = tid >> 4;        // 0..31
            const int kc = (tid & 15) * 2;  // 0,2,...,30
            const float2 f = *(const float2*)(x + (size_t)(row0 + r) * D_MODEL + kk + kc);
            v2bf pk; pk[0] = (__bf16)f.x; pk[1] = (__bf16)f.y;
            *(v2bf*)(Abuf + r * 40 + kc) = pk;
            if (kk + 32 < D_MODEL)
                __builtin_prefetch(x + (size_t)(row0 + r) * D_MODEL + kk + 32 + kc, 0, 1);
        }
        __syncthreads();

        v16bf afrag[2];
#pragma unroll
        for (int r = 0; r < 2; ++r) afrag[r] = load_afrag(Abuf, 40, lane, r, 0);

        const int kstep = kk >> 5;
#pragma unroll
        for (int j = 0; j < 3; ++j)
#pragma unroll
            for (int c = 0; c < 2; ++c) {
                const int t = wave * 2 + c;       // col tile 0..31
                v16bf bfrag = load_bfrag(W1ps[j] + ((size_t)(kstep * 32 + t) * 32 + lane) * 16);
#pragma unroll
                for (int r = 0; r < 2; ++r)
                    acc[j][r][c] = __builtin_amdgcn_wmma_f32_16x16x32_bf16(
                        false, afrag[r], false, bfrag, (short)0, acc[j][r][c], false, false);
            }
        __syncthreads();
    }

    // ================= per-MLP: GELU -> hidden -> GEMM2 =================
#pragma unroll
    for (int j = 0; j < 3; ++j) {
        if (j > 0) __syncthreads();   // previous GEMM2 done with hidden/w2lds

#if HAVE_TDM
        if (wave == 0)                 // kick W2 fragment DMA; overlaps epilogue
            tdm_load_16kb(W2ps[j], (unsigned)(unsigned long long)(uintptr_t)w2lds);
#endif
        // ---- epilogue 1: bias + exact GELU -> hidden LDS (bf16) ----
        {
            const int mofs = (lane < 16) ? 0 : 8;
#pragma unroll
            for (int c = 0; c < 2; ++c) {
                const int n  = wave * 32 + c * 16 + (lane & 15);
                const float bb = b1s[j][n];
#pragma unroll
                for (int r = 0; r < 2; ++r)
#pragma unroll
                    for (int e = 0; e < 8; ++e) {
                        const int row = r * 16 + mofs + e;
                        const float v = acc[j][r][c][e] + bb;
                        const float g = 0.5f * v * (1.0f + erff(v * 0.70710678118654752f));
                        hidden[row * 520 + n] = (__bf16)g;
                    }
            }
        }
#if HAVE_TDM
        if (wave == 0) __builtin_amdgcn_s_wait_tensorcnt(0);
#else
        {   // cooperative fallback copy of the 16 KB W2 fragment buffer
            const unsigned* src = (const unsigned*)W2ps[j];
            unsigned* dst = (unsigned*)w2lds;
            for (int i = tid; i < 4096; i += 512) dst[i] = src[i];
        }
#endif
        __syncthreads();

        // ---- GEMM2 on tensor cores: hidden[32x512] @ W2[512x16] ----
        if (wave < 2) {
            const int NOUT = (j == 0) ? 1 : 16;
            v8f acc2;
#pragma unroll
            for (int i = 0; i < 8; ++i) acc2[i] = 0.0f;
#pragma unroll
            for (int ks = 0; ks < 16; ++ks) {
                v16bf a = load_afrag(hidden, 520, lane, wave, ks * 32);
                v16bf b = load_bfrag(w2lds + ((size_t)(ks * 32) + lane) * 16);
                acc2 = __builtin_amdgcn_wmma_f32_16x16x32_bf16(
                    false, a, false, b, (short)0, acc2, false, false);
            }
            // ---- epilogue 2: activation + gate scaling ----
            const int n    = lane & 15;
            const int mofs = (lane < 16) ? 0 : 8;
            const float bv = (n < NOUT) ? b2s[j][n] : 0.0f;
#pragma unroll
            for (int e = 0; e < 8; ++e) {
                const int row = wave * 16 + mofs + e;
                const size_t rg = (size_t)(row0 + row) * D_STATE;
                const float s = acc2[e] + bv;
                if (j == 0) {
                    if (n == 0) {
                        const float d = (s > 20.0f) ? s : log1pf(expf(s)); // softplus
#pragma unroll
                        for (int n2 = 0; n2 < D_STATE; ++n2)
                            outs[0][rg + n2] = expf(d * auxs[0][n2 * D_STATE + n2]);
                    }
                } else if (j == 1) {
                    outs[1][rg + n] = tanhf(s) * auxs[1][n];
                } else {
                    outs[2][rg + n] = auxs[2][n] / (1.0f + expf(-s));
                }
            }
        }
    }
}

// ---------------------------------------------------------------------------
// Selective scan: h[b,d,:] independent per (b,d); one thread per d, 128 d's
// per block, grid (D/128, B). Gate rows are uniform per block (scalar
// broadcast); x loads coalesced; prefetch ahead on the x stream.
// ---------------------------------------------------------------------------
__global__ __launch_bounds__(128)
void scan_kernel(const float* __restrict__ x,
                 const float* __restrict__ decay,
                 const float* __restrict__ Bt,
                 const float* __restrict__ Ct,
                 float* __restrict__ y)
{
    const int d = blockIdx.x * 128 + threadIdx.x;
    const int b = blockIdx.y;
    const float* xp = x + (size_t)b * SEQ * D_MODEL + d;
    float*       yp = y + (size_t)b * SEQ * D_MODEL + d;
    const float4* dp = (const float4*)(decay + (size_t)b * SEQ * D_STATE);
    const float4* bp = (const float4*)(Bt    + (size_t)b * SEQ * D_STATE);
    const float4* cp = (const float4*)(Ct    + (size_t)b * SEQ * D_STATE);

    float h[D_STATE];
#pragma unroll
    for (int n = 0; n < D_STATE; ++n) h[n] = 0.0f;

    for (int s = 0; s < SEQ; ++s) {
        const float xv = xp[(size_t)s * D_MODEL];
        if (s + 8 < SEQ)
            __builtin_prefetch(xp + (size_t)(s + 8) * D_MODEL, 0, 1);

        float dv[16], bv[16], cv[16];
#pragma unroll
        for (int q = 0; q < 4; ++q) {
            const float4 dq = dp[s * 4 + q];
            const float4 bq = bp[s * 4 + q];
            const float4 cq = cp[s * 4 + q];
            dv[q*4+0]=dq.x; dv[q*4+1]=dq.y; dv[q*4+2]=dq.z; dv[q*4+3]=dq.w;
            bv[q*4+0]=bq.x; bv[q*4+1]=bq.y; bv[q*4+2]=bq.z; bv[q*4+3]=bq.w;
            cv[q*4+0]=cq.x; cv[q*4+1]=cq.y; cv[q*4+2]=cq.z; cv[q*4+3]=cq.w;
        }
        float acc = 0.0f;
#pragma unroll
        for (int n = 0; n < D_STATE; ++n) {
            h[n] = h[n] * dv[n] + bv[n] * xv;
            acc += h[n] * cv[n];
        }
        yp[(size_t)s * D_MODEL] = acc;
    }
}

// ---------------------------------------------------------------------------
// Launcher. Input order: x, A, Bp, Cp, dW1, db1, dW2, db2,
//                        bW1, bb1, bW2, bb2, cW1, cb1, cW2, cb2
// Workspace (~10 MB):
//   [0,3MB)          packed W1 bf16 x3 (1 MB each)
//   [3MB,3MB+96KB)   packed W2 bf16 x3 (16 KB each, 32 KB stride)
//   [4,6MB) decay    [6,8MB) Bt    [8,10MB) Ct      (f32 [32768][16])
// ---------------------------------------------------------------------------
extern "C" void kernel_launch(void* const* d_in, const int* in_sizes, int n_in,
                              void* d_out, int out_size, void* d_ws, size_t ws_size,
                              hipStream_t stream) {
    const float* x   = (const float*)d_in[0];
    const float* A   = (const float*)d_in[1];
    const float* Bp  = (const float*)d_in[2];
    const float* Cp  = (const float*)d_in[3];
    const float* dW1 = (const float*)d_in[4];
    const float* db1 = (const float*)d_in[5];
    const float* dW2 = (const float*)d_in[6];
    const float* db2 = (const float*)d_in[7];
    const float* bW1 = (const float*)d_in[8];
    const float* bb1 = (const float*)d_in[9];
    const float* bW2 = (const float*)d_in[10];
    const float* bb2 = (const float*)d_in[11];
    const float* cW1 = (const float*)d_in[12];
    const float* cb1 = (const float*)d_in[13];
    const float* cW2 = (const float*)d_in[14];
    const float* cb2 = (const float*)d_in[15];
    float* y = (float*)d_out;

    char* ws = (char*)d_ws;
    const size_t MB = 1u << 20;
    __bf16* w1p_d = (__bf16*)(ws + 0 * MB);
    __bf16* w1p_b = (__bf16*)(ws + 1 * MB);
    __bf16* w1p_c = (__bf16*)(ws + 2 * MB);
    __bf16* w2p_d = (__bf16*)(ws + 3 * MB);
    __bf16* w2p_b = (__bf16*)(ws + 3 * MB + 32 * 1024);
    __bf16* w2p_c = (__bf16*)(ws + 3 * MB + 64 * 1024);
    float*  decay = (float*)(ws + 4 * MB);
    float*  Btv   = (float*)(ws + 6 * MB);
    float*  Ctv   = (float*)(ws + 8 * MB);

    // 1) pack weights into WMMA fragment layouts
    const int packBlocks = (D_MODEL * D_HID) / 256;   // 2048
    pack_w1<<<packBlocks, 256, 0, stream>>>(dW1, w1p_d);
    pack_w1<<<packBlocks, 256, 0, stream>>>(bW1, w1p_b);
    pack_w1<<<packBlocks, 256, 0, stream>>>(cW1, w1p_c);
    pack_w2<<<32, 256, 0, stream>>>(dW2, w2p_d, 1);
    pack_w2<<<32, 256, 0, stream>>>(bW2, w2p_b, 16);
    pack_w2<<<32, 256, 0, stream>>>(cW2, w2p_c, 16);

    // 2) fused triple MLP -> decay / Bt / Ct  (32768 rows / 32 per block)
    mlp3_fused<<<ROWS / 32, 512, 0, stream>>>(
        x, w1p_d, w1p_b, w1p_c, db1, bb1, cb1,
        w2p_d, w2p_b, w2p_c, db2, bb2, cb2,
        A, Bp, Cp, decay, Btv, Ctv);

    // 3) sequential selective scan -> y
    dim3 grid(D_MODEL / 128, BATCH);
    scan_kernel<<<grid, 128, 0, stream>>>(x, decay, Btv, Ctv, y);
}
